// SelfAttentionLayer_6167573037767
// MI455X (gfx1250) — matrix operand order repaired and checked by reading, not compiled, not run on hardware
//
#include <hip/hip_runtime.h>
#include <hip/hip_bf16.h>

// ---------------------------------------------------------------------------
// CDNA5 (gfx1250) self-attention: bf16 WMMA everywhere, fp32 accumulate.
// ---------------------------------------------------------------------------

typedef __attribute__((ext_vector_type(16))) __bf16 bf16x16;
typedef __attribute__((ext_vector_type(8)))  float  v8f;

// 8 VGPRs = 16 bf16 elements per lane: one WMMA A/B fragment.
struct Frag {
    union {
        bf16x16 bf;
        uint4   q[2];
    };
};

__device__ __forceinline__ unsigned short f2bf(float f) {
    union { float f; unsigned u; } v;
    v.f = f;
    unsigned r = v.u + 0x7FFFu + ((v.u >> 16) & 1u);  // round-to-nearest-even
    return (unsigned short)(r >> 16);
}

__device__ __forceinline__ void load_frag(Frag& fr, const unsigned short* p0,
                                          const unsigned short* p1) {
    fr.q[0] = *reinterpret_cast<const uint4*>(p0);
    fr.q[1] = *reinterpret_cast<const uint4*>(p1);
}

__device__ __forceinline__ v8f wmma_bf16(const Frag& a, const Frag& b, v8f c) {
    return __builtin_amdgcn_wmma_f32_16x16x32_bf16(
        /*neg_a=*/false, a.bf, /*neg_b=*/false, b.bf,
        /*c_mod=*/(short)0, c, /*reuse_a=*/false, /*reuse_b=*/false);
}

// ---------------------------------------------------------------------------
// Conversion kernels
// ---------------------------------------------------------------------------
__global__ void cvt_f32_to_bf16(const float* __restrict__ in,
                                unsigned short* __restrict__ out, int n) {
    int i = blockIdx.x * blockDim.x + threadIdx.x;
    if (i < n) out[i] = f2bf(in[i]);
}

// W (K x N, row-major f32) -> Wt (N x K, row-major bf16)
__global__ void transpose_w(const float* __restrict__ w,
                            unsigned short* __restrict__ wt, int K, int N) {
    int i = blockIdx.x * blockDim.x + threadIdx.x;
    if (i < N * K) {
        int n = i / K, k = i - n * K;
        wt[i] = f2bf(w[(size_t)k * N + n]);
    }
}

// ---------------------------------------------------------------------------
// GEMM: C(8192x512) = A(8192x512,bf16) * B, with B supplied as Bt(512x512,bf16 NxK).
// Workgroup tile 128(M) x 128(N): 8 waves, each wave 32x64 (8 C fragments,
// 8 WMMAs per K-step). Fully unrolled K loop with explicit double buffering:
// all 12 loads of step k+1 issue before the 8 WMMAs of step k.
// mode 0: write bf16 into (B,H,S,DH) head layout   (Q, K projections)
// mode 1: write bf16 into (B,H,DH,S) transposed    (V projection)
// mode 2: write f32  into plain MxN                (output projection)
// ---------------------------------------------------------------------------
constexpr int GK = 512;   // inner dimension (= D)
constexpr int GN = 512;   // output columns (= D)

__global__ __launch_bounds__(256) void gemm_bf16(
    const unsigned short* __restrict__ A,
    const unsigned short* __restrict__ Bt,
    void* __restrict__ out, int mode) {
    const int lane = threadIdx.x & 31;
    const int wave = threadIdx.x >> 5;
    const int wm = wave & 3, wn = wave >> 2;
    const int m0 = blockIdx.x * 128 + wm * 32;
    const int n0 = blockIdx.y * 128 + wn * 64;
    const int lr = lane & 15;
    const int hi = lane >> 4;
    const int achunk = hi ? 8 : 0;    // A frag: K chunks {c, c+16}, c in {0,8}
    const int bchunk = hi ? 8 : 0;    // B frag: K chunks {c, c+16}, c in {0,8}

    const unsigned short* ar0 = A + (size_t)(m0 + lr) * GK + achunk;
    const unsigned short* ar1 = A + (size_t)(m0 + 16 + lr) * GK + achunk;
    const unsigned short* br  = Bt + (size_t)(n0 + lr) * GK + bchunk;

    v8f acc[8] = {};
    Frag a[2], b[4];
    load_frag(a[0], ar0, ar0 + 16);
    load_frag(a[1], ar1, ar1 + 16);
#pragma unroll
    for (int f = 0; f < 4; ++f) {
        const unsigned short* bp = br + (size_t)f * 16 * GK;
        load_frag(b[f], bp, bp + 16);
    }

#pragma unroll
    for (int k0 = 0; k0 < GK; k0 += 32) {
        // prefetch next K-step fragments (dummy reload of k=0 on the last step)
        const int kn = (k0 + 32 < GK) ? k0 + 32 : 0;
        Frag an[2], bn[4];
        load_frag(an[0], ar0 + kn, ar0 + kn + 16);
        load_frag(an[1], ar1 + kn, ar1 + kn + 16);
#pragma unroll
        for (int f = 0; f < 4; ++f) {
            const unsigned short* bp = br + (size_t)f * 16 * GK + kn;
            load_frag(bn[f], bp, bp + 16);
        }
        // 8 back-to-back WMMAs on the current buffers
#pragma unroll
        for (int ai = 0; ai < 2; ++ai)
#pragma unroll
            for (int f = 0; f < 4; ++f)
                acc[ai * 4 + f] = wmma_bf16(a[ai], b[f], acc[ai * 4 + f]);
        // rotate (pure renames under full unroll)
#pragma unroll
        for (int i = 0; i < 2; ++i) a[i] = an[i];
#pragma unroll
        for (int f = 0; f < 4; ++f) b[f] = bn[f];
    }

#pragma unroll
    for (int ai = 0; ai < 2; ++ai) {
#pragma unroll
        for (int f = 0; f < 4; ++f) {
#pragma unroll
            for (int i = 0; i < 8; ++i) {
                const int m = m0 + ai * 16 + i + hi * 8;
                const int n = n0 + f * 16 + lr;
                const float v = acc[ai * 4 + f][i];
                if (mode == 2) {
                    ((float*)out)[(size_t)m * GN + n] = v;
                } else {
                    const int bb = m >> 11;        // m / 2048
                    const int s  = m & 2047;
                    const int h  = n >> 6;         // n / 64
                    const int dh = n & 63;
                    const unsigned short bv = f2bf(v);
                    if (mode == 0)
                        ((unsigned short*)out)[(((size_t)(bb * 8 + h) * 2048) + s) * 64 + dh] = bv;
                    else
                        ((unsigned short*)out)[(((size_t)(bb * 8 + h) * 64) + dh) * 2048 + s] = bv;
                }
            }
        }
    }
}

// ---------------------------------------------------------------------------
// Flash attention: one wave per 16-query tile per (b,h).  Single-wave
// workgroups keep the per-q-tile causal trip count barrier-safe (barriers
// lower to S_NOP).  V fragments for the current block and K fragments for
// the NEXT block are issued before the softmax so global latency overlaps
// the exp/shfl work.
// Q,K in (B,H,S,DH) bf16; Vt in (B,H,DH,S) bf16; Ctx out (B,S,D) bf16.
// ---------------------------------------------------------------------------
__global__ __launch_bounds__(32) void attn_flash(
    const unsigned short* __restrict__ Q,
    const unsigned short* __restrict__ Kh,
    const unsigned short* __restrict__ Vt,
    const int* __restrict__ tmask,
    unsigned short* __restrict__ Ctx) {
    __shared__ unsigned short pbuf[16 * 32];   // P tile staging (C-layout -> A-layout)

    const int lane = threadIdx.x & 31;
    const int lr = lane & 15;
    const int hi = lane >> 4;
    const int achunk = hi ? 16 : 0;
    const int bchunk = hi ? 16 : 0;

    int id = blockIdx.x;
    const int qt = id & 127; id >>= 7;
    const int h  = id & 7;
    const int b  = id >> 3;
    const int q0 = qt * 16;
    const size_t head = (size_t)(b * 8 + h) * 2048 * 64;

    // Q A-fragments for dh 0..31 and 32..63
    Frag qa0, qa1;
    {
        const unsigned short* qrow = Q + head + (size_t)(q0 + lr) * 64;
        load_frag(qa0, qrow + achunk,      qrow + achunk + 8);
        load_frag(qa1, qrow + 32 + achunk, qrow + 32 + achunk + 8);
    }

    auto load_kb = [&](Frag* kf, int j0) {
        const unsigned short* k0p = Kh + head + (size_t)(j0 + lr) * 64 + bchunk;
        const unsigned short* k1p = k0p + 16 * 64;
        load_frag(kf[0], k0p,      k0p + 8);
        load_frag(kf[1], k0p + 32, k0p + 40);
        load_frag(kf[2], k1p,      k1p + 8);
        load_frag(kf[3], k1p + 32, k1p + 40);
    };

    float rm[8], rs[8];
#pragma unroll
    for (int i = 0; i < 8; ++i) { rm[i] = -1e38f; rs[i] = 0.f; }
    v8f acc[4] = {};

    const int jend = q0 + 16;                 // causal: keys < q0+16 needed
    Frag kb[4];
    load_kb(kb, 0);

    for (int j0 = 0; j0 < jend; j0 += 32) {
        // ---- scores: two 16x16 tiles over keys [j0,j0+16) and [j0+16,j0+32)
        v8f c0 = {}, c1 = {};
        c0 = wmma_bf16(qa0, kb[0], c0);
        c0 = wmma_bf16(qa1, kb[1], c0);
        c1 = wmma_bf16(qa0, kb[2], c1);
        c1 = wmma_bf16(qa1, kb[3], c1);

        // ---- issue V loads (this block) and K loads (next block) now, so
        //      their latency hides under the softmax transcendentals
        Frag vb[4];
#pragma unroll
        for (int f = 0; f < 4; ++f) {
            const unsigned short* vp =
                Vt + head + (size_t)(f * 16 + lr) * 2048 + j0 + bchunk;
            load_frag(vb[f], vp, vp + 8);
        }
        Frag kn[4];
        const int jn = (j0 + 32 < jend) ? j0 + 32 : 0;   // dummy on last step
        load_kb(kn, jn);

        const int key0 = j0 + lr, key1 = key0 + 16;
        const int tk0 = tmask[b * 2048 + key0];
        const int tk1 = tmask[b * 2048 + key1];

        float corr[8];
#pragma unroll
        for (int i = 0; i < 8; ++i) {
            const int row  = i + hi * 8;
            const int qidx = q0 + row;
            float s0 = c0[i] * 0.125f;        // 1/sqrt(64)
            float s1 = c1[i] * 0.125f;
            if (key0 > qidx || tk0 == 0) s0 = -1e30f;
            if (key1 > qidx || tk1 == 0) s1 = -1e30f;
            // half-wave row reductions (row elements live across 16 lanes)
            float tm = fmaxf(s0, s1);
            tm = fmaxf(tm, __shfl_xor(tm, 1, 32));
            tm = fmaxf(tm, __shfl_xor(tm, 2, 32));
            tm = fmaxf(tm, __shfl_xor(tm, 4, 32));
            tm = fmaxf(tm, __shfl_xor(tm, 8, 32));
            const float nm = fmaxf(rm[i], tm);
            const float cr = __expf(rm[i] - nm);
            const float p0 = (s0 <= -1e29f) ? 0.f : __expf(s0 - nm);
            const float p1 = (s1 <= -1e29f) ? 0.f : __expf(s1 - nm);
            float ps = p0 + p1;
            ps += __shfl_xor(ps, 1, 32);
            ps += __shfl_xor(ps, 2, 32);
            ps += __shfl_xor(ps, 4, 32);
            ps += __shfl_xor(ps, 8, 32);
            rs[i] = rs[i] * cr + ps;
            rm[i] = nm;
            corr[i] = cr;
            pbuf[row * 32 + lr]      = f2bf(p0);
            pbuf[row * 32 + lr + 16] = f2bf(p1);
        }
#pragma unroll
        for (int f = 0; f < 4; ++f)
#pragma unroll
            for (int i = 0; i < 8; ++i) acc[f][i] *= corr[i];

        __syncthreads();                       // LDS stores -> A-fragment loads

        Frag pa;                               // P as 16x32 A-fragment
        {
            const unsigned short* pp = pbuf + lr * 32 + achunk;
            load_frag(pa, pp, pp + 8);
        }
#pragma unroll
        for (int f = 0; f < 4; ++f)
            acc[f] = wmma_bf16(pa, vb[f], acc[f]);

        __syncthreads();
#pragma unroll
        for (int f = 0; f < 4; ++f) kb[f] = kn[f];
    }

    // ---- normalize and emit context in (B,S,D) bf16 for the output GEMM
#pragma unroll
    for (int i = 0; i < 8; ++i) {
        const int row  = i + hi * 8;
        const int qidx = q0 + row;
        const int tq = tmask[b * 2048 + qidx];
        const float inv = (tq != 0 && rs[i] > 0.f) ? 1.f / rs[i] : 0.f;
#pragma unroll
        for (int f = 0; f < 4; ++f) {
            Ctx[((size_t)(b * 2048) + qidx) * 512 + h * 64 + f * 16 + lr] =
                f2bf(acc[f][i] * inv);
        }
    }
}

// ---------------------------------------------------------------------------
// Host launcher
// ---------------------------------------------------------------------------
extern "C" void kernel_launch(void* const* d_in, const int* in_sizes, int n_in,
                              void* d_out, int out_size, void* d_ws, size_t ws_size,
                              hipStream_t stream) {
    constexpr int B = 4, S = 2048, D = 512, H = 8;
    constexpr int M = B * S;                  // 8192

    const float* emb   = (const float*)d_in[0];
    const int*   tmask = (const int*)d_in[1];
    const float* Wq    = (const float*)d_in[2];
    const float* Wk    = (const float*)d_in[3];
    const float* Wv    = (const float*)d_in[4];
    const float* Wo    = (const float*)d_in[5];

    char* ws = (char*)d_ws;
    unsigned short* Ebf = (unsigned short*)ws; ws += (size_t)M * D * 2;
    unsigned short* WqT = (unsigned short*)ws; ws += (size_t)D * D * 2;
    unsigned short* WkT = (unsigned short*)ws; ws += (size_t)D * D * 2;
    unsigned short* WvT = (unsigned short*)ws; ws += (size_t)D * D * 2;
    unsigned short* WoT = (unsigned short*)ws; ws += (size_t)D * D * 2;
    unsigned short* Qbf = (unsigned short*)ws; ws += (size_t)M * D * 2;
    unsigned short* Kbf = (unsigned short*)ws; ws += (size_t)M * D * 2;
    unsigned short* Vtb = (unsigned short*)ws; ws += (size_t)M * D * 2;
    unsigned short* Ctx = (unsigned short*)ws; ws += (size_t)M * D * 2;

    // 1) convert embeddings + transpose/convert weights to bf16
    const int nE = M * D;
    cvt_f32_to_bf16<<<(nE + 255) / 256, 256, 0, stream>>>(emb, Ebf, nE);
    const int nW = D * D;
    transpose_w<<<(nW + 255) / 256, 256, 0, stream>>>(Wq, WqT, D, D);
    transpose_w<<<(nW + 255) / 256, 256, 0, stream>>>(Wk, WkT, D, D);
    transpose_w<<<(nW + 255) / 256, 256, 0, stream>>>(Wv, WvT, D, D);
    transpose_w<<<(nW + 255) / 256, 256, 0, stream>>>(Wo, WoT, D, D);

    // 2) QKV projections (Q,K -> head layout; V -> transposed head layout)
    dim3 g(M / 128, D / 128);                 // (64, 4)
    gemm_bf16<<<g, 256, 0, stream>>>(Ebf, WqT, Qbf, 0);
    gemm_bf16<<<g, 256, 0, stream>>>(Ebf, WkT, Kbf, 0);
    gemm_bf16<<<g, 256, 0, stream>>>(Ebf, WvT, Vtb, 1);

    // 3) masked causal flash attention
    attn_flash<<<B * H * (S / 16), 32, 0, stream>>>(Qbf, Kbf, Vtb, tmask, Ctx);

    // 4) output projection -> fp32 d_out
    gemm_bf16<<<g, 256, 0, stream>>>(Ctx, WoT, d_out, 2);
}
